// scTransformerLayer_86543591014688
// MI455X (gfx1250) — matrix operand order repaired and checked by reading, not compiled, not run on hardware
//
#include <hip/hip_runtime.h>
#include <hip/hip_bf16.h>
#include <cstdint>
#include <cstddef>

// ---------------------------------------------------------------------------
// Problem constants (from the reference)
// ---------------------------------------------------------------------------
#define DMODEL 1024
#define NHEADS 16
#define HDIM   64
#define DFF    4096
#define NB     4
#define LSEQ   4096
#define SSEQ   4096
#define STT    (LSEQ + SSEQ)      // 8192 : concatenated KV sequence length

typedef __bf16 bf16_t;
typedef __bf16 v2bf  __attribute__((ext_vector_type(2)));
typedef __bf16 v8bf  __attribute__((ext_vector_type(8)));
typedef __bf16 v16bf __attribute__((ext_vector_type(16)));
typedef float  v8f   __attribute__((ext_vector_type(8)));

// Tensor Data Mover support (probe-confirmed builtin on ROCm 7.2 clang-22;
// therock clang-23 uses a 6-arg form). Falls back to register-staged loads.
#if defined(__has_builtin)
#if __has_builtin(__builtin_amdgcn_tensor_load_to_lds) && \
    __has_builtin(__builtin_amdgcn_s_wait_tensorcnt)
#define ATHENA_USE_TDM 1
#endif
#endif
#ifndef ATHENA_USE_TDM
#define ATHENA_USE_TDM 0
#endif

#if ATHENA_USE_TDM
typedef uint32_t u32x4 __attribute__((ext_vector_type(4)));
typedef int      i32x4 __attribute__((ext_vector_type(4)));
typedef int      i32x8 __attribute__((ext_vector_type(8)));
typedef __attribute__((address_space(3))) bf16_t lds_bf16_t;

// Issue one TDM load of a (tile_k=32 x tile_m=128) bf16 tile (row stride K
// elements) into LDS at lds_off, with pad: +4 DWORDs after every 16 DWORDs
// (i.e., 32 bf16 row -> 40 bf16 LDS stride, matching As[.][128][40]).
static __device__ __forceinline__ void tdm_load_tile_a(
    const bf16_t* gptr, uint32_t lds_off, int K) {
  const uint64_t ga = (uint64_t)(uintptr_t)gptr;
  u32x4 g0;
  g0[0] = 1u;                                        // count=1, user mode
  g0[1] = lds_off;                                   // lds_addr (bytes)
  g0[2] = (uint32_t)ga;                              // global_addr[31:0]
  g0[3] = (uint32_t)((ga >> 32) & 0x1FFFFFFu) | (2u << 30);  // addr hi | type=2
  i32x8 g1;
  // data_size=2B | pad_enable | pad_interval=16DW(code 3) | pad_amount=4DW(code 3)
  g1[0] = (1 << 16) | (1 << 20) | (3 << 22) | (3 << 25);
  g1[1] = (int)0xFFFF0000u;          // tensor_dim0 lo16 (huge: no OOB clamp)
  g1[2] = (int)0xFFFF7FFFu;          // tensor_dim0 hi16 | tensor_dim1 lo16
  g1[3] = (int)((32u << 16) | 0x7FFFu);  // tensor_dim1 hi16 | tile_dim0=32
  g1[4] = 128;                       // tile_dim1=128 (tile_dim2=0)
  g1[5] = K;                         // tensor_dim0_stride[31:0] (elements)
  g1[6] = 0;
  g1[7] = 0;
  i32x4 gz = {0, 0, 0, 0};
#if __clang_major__ >= 23
  i32x8 gz8 = {0, 0, 0, 0, 0, 0, 0, 0};
  __builtin_amdgcn_tensor_load_to_lds(g0, g1, gz, gz, gz8, 0);
#else
  __builtin_amdgcn_tensor_load_to_lds(g0, g1, gz, gz, 0);
#endif
}
#endif  // ATHENA_USE_TDM

static __device__ __forceinline__ v16bf make_frag(v8bf lo, v8bf hi) {
  v16bf r;
#pragma unroll
  for (int i = 0; i < 8; ++i) { r[i] = lo[i]; r[i + 8] = hi[i]; }
  return r;
}

static __device__ __forceinline__ v8f vzero8() {
  v8f z;
#pragma unroll
  for (int e = 0; e < 8; ++e) z[e] = 0.0f;
  return z;
}

// ---------------------------------------------------------------------------
// Elementwise helpers
// ---------------------------------------------------------------------------
__global__ void f2bf_kernel(const float* __restrict__ x, bf16_t* __restrict__ y,
                            size_t nElem) {
  size_t i = (size_t)blockIdx.x * blockDim.x + threadIdx.x;
  if (i < nElem) y[i] = (bf16_t)x[i];
}

// Build bf16 [kv_src ; q_src] along the sequence dim: (N, STT, DMODEL)
__global__ void kvcat_kernel(const float* __restrict__ kv,
                             const float* __restrict__ q,
                             bf16_t* __restrict__ out) {
  size_t i = (size_t)blockIdx.x * blockDim.x + threadIdx.x;  // NB*STT*DMODEL
  const int    c = (int)(i & (DMODEL - 1));
  const size_t t = i >> 10;                 // (n, s)
  const int    s = (int)(t & (STT - 1));
  const int    n = (int)(t >> 13);
  float v = (s < SSEQ)
                ? kv[(((size_t)n * SSEQ + s) << 10) + c]
                : q[(((size_t)n * LSEQ + (s - SSEQ)) << 10) + c];
  out[i] = (bf16_t)v;
}

// ---------------------------------------------------------------------------
// Generic bf16 WMMA GEMM:  C(MxN) = epi( A(MxK) @ B(KxN) + bias )
//   Block tile 128x128, 256 threads = 8 waves (2x4), wave tile 64x32.
//   Double-buffered LDS pipeline: while WMMA consumes buffer t&1, the TDM
//   (or register-staged fallback) fills buffer t^1 for the next k-step;
//   one s_wait_tensorcnt + one workgroup barrier per k-step.
//   epi: 0 = none, 1 = elu(x)+1, 2 = x*scale, 3 = relu(x)
// ---------------------------------------------------------------------------
__global__ __launch_bounds__(256) void gemm_bf16_kernel(
    const bf16_t* __restrict__ A, const bf16_t* __restrict__ B,
    const float* __restrict__ bias, void* __restrict__ Cout,
    int M, int Ndim, int K, int epi, float scale, int bf16out) {
  __shared__ bf16_t As[2][128][40];   // (m, k) row-major, padded (TDM pad)
  __shared__ bf16_t Bs[2][128][40];   // (n, k) transposed,  padded

  const int tid  = threadIdx.x;
  const int wave = tid >> 5;
  const int lane = tid & 31;
  const int half = lane >> 4;
  const int l16  = lane & 15;
  const int wm   = wave >> 2;   // 0..1 -> 64-row strip
  const int wn   = wave & 3;    // 0..3 -> 32-col strip

  const int bm0 = blockIdx.y * 128;
  const int bn0 = blockIdx.x * 128;

  v8f acc[4][2];
#pragma unroll
  for (int i = 0; i < 4; ++i)
#pragma unroll
    for (int j = 0; j < 2; ++j) acc[i][j] = vzero8();

#if !ATHENA_USE_TDM
  const int arow = tid >> 1;          // 0..127
  const int acol = (tid & 1) << 4;    // 0 / 16
#else
  const uint32_t As_lds = (uint32_t)(uintptr_t)(lds_bf16_t*)&As[0][0][0];
#endif
  const int brow = (tid >> 4) << 1;   // 0,2,..,30 (k pair)
  const int bcol = (tid & 15) << 3;   // 0..120    (n base)

  // Stage the A tile for k-offset k0 into LDS buffer `buf`.
  auto stage_A = [&](int k0, int buf) {
#if ATHENA_USE_TDM
    if (wave == 0)
      tdm_load_tile_a(A + (size_t)bm0 * K + k0,
                      As_lds + (uint32_t)buf * (uint32_t)sizeof(As[0]), K);
#else
    const bf16_t* Ag = A + (size_t)(bm0 + arow) * K + (k0 + acol);
    v8bf a0 = *(const v8bf*)Ag;
    v8bf a1 = *(const v8bf*)(Ag + 8);
    *(v8bf*)&As[buf][arow][acol]     = a0;
    *(v8bf*)&As[buf][arow][acol + 8] = a1;
#endif
  };
  // Stage (transpose) the B tile for k-offset k0 into LDS buffer `buf`.
  auto stage_B = [&](int k0, int buf) {
    const bf16_t* Bg = B + (size_t)(k0 + brow) * Ndim + (bn0 + bcol);
    v8bf b0 = *(const v8bf*)Bg;          // row k0+brow
    v8bf b1 = *(const v8bf*)(Bg + Ndim); // row k0+brow+1
    if (k0 + 64 < K)                     // pull the tile after next toward L2
      __builtin_prefetch(Bg + (size_t)64 * Ndim, 0, 1);
#pragma unroll
    for (int j = 0; j < 8; ++j) {        // packed (k, k+1) pair per n
      v2bf pk;
      pk[0] = b0[j];
      pk[1] = b1[j];
      *(v2bf*)&Bs[buf][bcol + j][brow] = pk;  // 32-bit DS store
    }
  };

  // Prologue: fill buffer 0 with tile 0.
  stage_A(0, 0);
  stage_B(0, 0);
#if ATHENA_USE_TDM
  __builtin_amdgcn_s_wait_tensorcnt(0);
#endif
  __syncthreads();

  const int T = K >> 5;
  for (int kt = 0; kt < T; ++kt) {
    const int cur = kt & 1;
    if (kt + 1 < T) {            // overlap next-tile DMA/loads with compute
      stage_A((kt + 1) << 5, cur ^ 1);
      stage_B((kt + 1) << 5, cur ^ 1);
    }

    v16bf bfrag[2];
#pragma unroll
    for (int j = 0; j < 2; ++j) {
      const int nc = wn * 32 + j * 16 + l16;
      v8bf lo = *(const v8bf*)&Bs[cur][nc][16 * half];
      v8bf hi = *(const v8bf*)&Bs[cur][nc][16 * half + 8];
      bfrag[j] = make_frag(lo, hi);
    }
#pragma unroll
    for (int i = 0; i < 4; ++i) {
      const int mr = wm * 64 + i * 16 + l16;
      v8bf lo = *(const v8bf*)&As[cur][mr][16 * half];
      v8bf hi = *(const v8bf*)&As[cur][mr][16 * half + 8];
      v16bf af = make_frag(lo, hi);
#pragma unroll
      for (int j = 0; j < 2; ++j)
        acc[i][j] = __builtin_amdgcn_wmma_f32_16x16x32_bf16(
            false, af, false, bfrag[j], (short)0, acc[i][j], false, false);
    }
#if ATHENA_USE_TDM
    __builtin_amdgcn_s_wait_tensorcnt(0);  // next tile's DMA done (wave 0)
#endif
    __syncthreads();
  }

  // Epilogue: C layout -> c[e] = C[M = e + 8*half][N = l16] per 16x16 tile
#pragma unroll
  for (int i = 0; i < 4; ++i) {
    const int rowb = bm0 + wm * 64 + i * 16 + 8 * half;
#pragma unroll
    for (int j = 0; j < 2; ++j) {
      const int col = bn0 + wn * 32 + j * 16 + l16;
      const float bv = bias ? bias[col] : 0.0f;
#pragma unroll
      for (int e = 0; e < 8; ++e) {
        float v = acc[i][j][e] + bv;
        if (epi == 1)      v = (v > 0.0f) ? (v + 1.0f) : __expf(v);  // elu+1
        else if (epi == 2) v = v * scale;
        else if (epi == 3) v = fmaxf(v, 0.0f);
        const size_t idx = (size_t)(rowb + e) * Ndim + col;
        if (bf16out) ((bf16_t*)Cout)[idx] = (bf16_t)v;
        else         ((float*)Cout)[idx]  = v;
      }
    }
  }
}

// ---------------------------------------------------------------------------
// Per-head KV accumulation: KV[n,h] (64x64) = sum_s Kf[n,s,h,:]^T Vs[n,s,h,:]
// One block (8 waves) per (n,h); waves tiled 4(m) x 2(n) over the 64x64 out.
// Double-buffered: one barrier per 32-wide s-chunk.
// ---------------------------------------------------------------------------
__global__ __launch_bounds__(256) void kv_accum_kernel(
    const bf16_t* __restrict__ Kf, const bf16_t* __restrict__ Vs,
    bf16_t* __restrict__ KV) {
  const int nh = blockIdx.x;
  const int n = nh >> 4, h = nh & 15;
  __shared__ bf16_t KfT[2][64][40];   // (d, s)
  __shared__ bf16_t VsT[2][64][40];   // (v, s)

  const int tid = threadIdx.x, wave = tid >> 5, lane = tid & 31;
  const int half = lane >> 4, l16 = lane & 15;
  const int wm = wave >> 1, wn = wave & 1;   // 4 x 2

  v8f acc[2];
  acc[0] = vzero8(); acc[1] = vzero8();

  // Loader mapping: waves 0-3 stage Kf, waves 4-7 stage Vs; each thread
  // loads two adjacent s-rows of 8 d-values and stores packed 32-bit pairs.
  const int mtx = tid >> 7;            // 0: Kf, 1: Vs
  const int r   = tid & 127;
  const int sp  = (r >> 3) << 1;       // 0,2,..,30
  const int dB  = (r & 7) << 3;        // 0..56
  const bf16_t* src = mtx ? Vs : Kf;

  auto stage = [&](int s0, int buf) {
    const size_t base =
        ((size_t)(n * STT + s0 + sp) * NHEADS + h) * HDIM + dB;
    v8bf r0 = *(const v8bf*)(src + base);
    v8bf r1 = *(const v8bf*)(src + base + NHEADS * HDIM);
#pragma unroll
    for (int j = 0; j < 8; ++j) {
      v2bf pk;
      pk[0] = r0[j];
      pk[1] = r1[j];
      if (mtx) *(v2bf*)&VsT[buf][dB + j][sp] = pk;
      else     *(v2bf*)&KfT[buf][dB + j][sp] = pk;
    }
  };

  stage(0, 0);
  __syncthreads();

  const int T = STT / 32;   // 256
  for (int t = 0; t < T; ++t) {
    const int cur = t & 1;
    if (t + 1 < T) stage((t + 1) * 32, cur ^ 1);

    const int mr = wm * 16 + l16;
    v8bf alo = *(const v8bf*)&KfT[cur][mr][16 * half];
    v8bf ahi = *(const v8bf*)&KfT[cur][mr][16 * half + 8];
    v16bf af = make_frag(alo, ahi);
#pragma unroll
    for (int j = 0; j < 2; ++j) {
      const int nc = wn * 32 + j * 16 + l16;
      v8bf blo = *(const v8bf*)&VsT[cur][nc][16 * half];
      v8bf bhi = *(const v8bf*)&VsT[cur][nc][16 * half + 8];
      v16bf bfg = make_frag(blo, bhi);
      acc[j] = __builtin_amdgcn_wmma_f32_16x16x32_bf16(
          false, af, false, bfg, (short)0, acc[j], false, false);
    }
    __syncthreads();
  }
#pragma unroll
  for (int j = 0; j < 2; ++j) {
    const int col = wn * 32 + j * 16 + l16;
#pragma unroll
    for (int e = 0; e < 8; ++e) {
      const int row = wm * 16 + e + 8 * half;
      KV[((size_t)nh * 64 + row) * 64 + col] = (bf16_t)acc[j][e];
    }
  }
}

// Ksum[n,h,d] = sum_s Kf[n,s,h,d]
__global__ __launch_bounds__(256) void ksum_kernel(const bf16_t* __restrict__ Kf,
                                                   float* __restrict__ Ksum) {
  const int nh = blockIdx.x;
  const int n = nh >> 4, h = nh & 15;
  const int d = threadIdx.x & 63;
  const int part = threadIdx.x >> 6;   // 0..3
  float s = 0.0f;
  for (int si = part; si < STT; si += 4)
    s += (float)Kf[((size_t)(n * STT + si) * NHEADS + h) * HDIM + d];
  __shared__ float red[256];
  red[threadIdx.x] = s;
  __syncthreads();
  if (part == 0)
    Ksum[(size_t)nh * HDIM + d] = red[d] + red[64 + d] + red[128 + d] + red[192 + d];
}

// Z[n,l,h] = 1 / (dot(Qf[n,l,h,:], Ksum[n,h,:]) + eps)
__global__ __launch_bounds__(256) void z_kernel(const bf16_t* __restrict__ Qf,
                                                const float* __restrict__ Ksum,
                                                float* __restrict__ Z) {
  const size_t idx = (size_t)blockIdx.x * 256 + threadIdx.x;  // < NB*LSEQ*NHEADS
  const int    h  = (int)(idx & 15);
  const size_t nl = idx >> 4;
  const int    n  = (int)(nl >> 12);  // / LSEQ
  const bf16_t* q  = Qf + (nl * NHEADS + h) * HDIM;
  const float*  ks = Ksum + ((size_t)n * NHEADS + h) * HDIM;
  float acc = 0.0f;
#pragma unroll
  for (int d = 0; d < HDIM; ++d) acc += (float)q[d] * ks[d];
  Z[idx] = 1.0f / (acc + 1e-6f);
}

// ---------------------------------------------------------------------------
// att[n,l,h,:] = (Qf[n,l,h,:] @ KV[n,h]) * Z[n,l,h] * STT  -> bf16 (N,L,DMODEL)
// Block: 128 rows of L x full 64 cols; 8 waves tiled 4(m) x 2(n).
// ---------------------------------------------------------------------------
__global__ __launch_bounds__(256) void attn_out_kernel(
    const bf16_t* __restrict__ Qf, const bf16_t* __restrict__ KV,
    const float* __restrict__ Z, bf16_t* __restrict__ attO) {
  const int nh = blockIdx.y;
  const int n = nh >> 4, h = nh & 15;
  const int l0 = blockIdx.x * 128;

  __shared__ bf16_t Qs[128][72];   // (l, d)
  __shared__ bf16_t KVT[64][72];   // (v, d)

  const int tid = threadIdx.x, wave = tid >> 5, lane = tid & 31;
  const int half = lane >> 4, l16 = lane & 15;
  const int wm = wave >> 1, wn = wave & 1;   // 4 x 2 -> 32x32 per wave

  {  // Q tile 128x64 (contiguous, no transpose)
    const int ll = tid >> 1;
    const int dB = (tid & 1) * 32;
    const bf16_t* qg =
        Qf + ((size_t)(n * LSEQ + l0 + ll) * NHEADS + h) * HDIM + dB;
#pragma unroll
    for (int j = 0; j < 4; ++j)
      *(v8bf*)&Qs[ll][dB + j * 8] = *(const v8bf*)(qg + j * 8);
  }
  {  // KV 64x64 transposed to (v, d): two d-rows per thread, packed stores
    const int dp = (tid >> 3) << 1;    // 0,2,..,62
    const int vB = (tid & 7) << 3;     // 0..56
    const bf16_t* kg = KV + ((size_t)nh * 64 + dp) * 64 + vB;
    v8bf k0 = *(const v8bf*)kg;
    v8bf k1 = *(const v8bf*)(kg + 64);
#pragma unroll
    for (int j = 0; j < 8; ++j) {
      v2bf pk;
      pk[0] = k0[j];
      pk[1] = k1[j];
      *(v2bf*)&KVT[vB + j][dp] = pk;
    }
  }
  __syncthreads();

  v8f acc[2][2];
#pragma unroll
  for (int i = 0; i < 2; ++i)
#pragma unroll
    for (int j = 0; j < 2; ++j) acc[i][j] = vzero8();

#pragma unroll
  for (int ks = 0; ks < 2; ++ks) {   // K = 64 = 2 x 32
    v16bf bfrag[2];
#pragma unroll
    for (int j = 0; j < 2; ++j) {
      const int nc = wn * 32 + j * 16 + l16;
      v8bf lo = *(const v8bf*)&KVT[nc][ks * 32 + 16 * half];
      v8bf hi = *(const v8bf*)&KVT[nc][ks * 32 + 16 * half + 8];
      bfrag[j] = make_frag(lo, hi);
    }
#pragma unroll
    for (int i = 0; i < 2; ++i) {
      const int mr = wm * 32 + i * 16 + l16;
      v8bf lo = *(const v8bf*)&Qs[mr][ks * 32 + 16 * half];
      v8bf hi = *(const v8bf*)&Qs[mr][ks * 32 + 16 * half + 8];
      v16bf af = make_frag(lo, hi);
#pragma unroll
      for (int j = 0; j < 2; ++j)
        acc[i][j] = __builtin_amdgcn_wmma_f32_16x16x32_bf16(
            false, af, false, bfrag[j], (short)0, acc[i][j], false, false);
    }
  }

#pragma unroll
  for (int i = 0; i < 2; ++i) {
#pragma unroll
    for (int e = 0; e < 8; ++e) {
      const int l = l0 + wm * 32 + i * 16 + e + 8 * half;
      const float zf = Z[((size_t)n * LSEQ + l) * NHEADS + h] * (float)STT;
#pragma unroll
      for (int j = 0; j < 2; ++j) {
        const int v = wn * 32 + j * 16 + l16;
        attO[((size_t)(n * LSEQ + l)) * DMODEL + h * HDIM + v] =
            (bf16_t)(acc[i][j][e] * zf);
      }
    }
  }
}

// ---------------------------------------------------------------------------
// LayerNorm over last dim (C=1024) with fused residual: y = LN(X + R)*g + b
// Writes f32 always; bf16 copy optionally (for the next GEMM input).
// ---------------------------------------------------------------------------
__global__ __launch_bounds__(256) void ln_kernel(
    const float* __restrict__ X, const float* __restrict__ R,
    const float* __restrict__ g, const float* __restrict__ b,
    float* __restrict__ outF, bf16_t* __restrict__ outBF) {
  const size_t row = blockIdx.x;
  const float* xr = X + row * DMODEL;
  const float* rr = R + row * DMODEL;
  float x[4], s = 0.0f, s2 = 0.0f;
#pragma unroll
  for (int i = 0; i < 4; ++i) {
    const int c = threadIdx.x + i * 256;
    x[i] = xr[c] + rr[c];
    s += x[i];
    s2 += x[i] * x[i];
  }
  __shared__ float rs[256], rs2[256];
  rs[threadIdx.x] = s;
  rs2[threadIdx.x] = s2;
  __syncthreads();
  for (int off = 128; off > 0; off >>= 1) {
    if (threadIdx.x < off) {
      rs[threadIdx.x] += rs[threadIdx.x + off];
      rs2[threadIdx.x] += rs2[threadIdx.x + off];
    }
    __syncthreads();
  }
  const float mean = rs[0] * (1.0f / DMODEL);
  const float var  = rs2[0] * (1.0f / DMODEL) - mean * mean;
  const float inv  = rsqrtf(var + 1e-5f);
#pragma unroll
  for (int i = 0; i < 4; ++i) {
    const int c = threadIdx.x + i * 256;
    const float y = (x[i] - mean) * inv * g[c] + b[c];
    outF[row * DMODEL + c] = y;
    if (outBF) outBF[row * DMODEL + c] = (bf16_t)y;
  }
}

// ---------------------------------------------------------------------------
// Host orchestration
// ---------------------------------------------------------------------------
extern "C" void kernel_launch(void* const* d_in, const int* in_sizes, int n_in,
                              void* d_out, int out_size, void* d_ws,
                              size_t ws_size, hipStream_t stream) {
  (void)in_sizes; (void)n_in; (void)out_size; (void)ws_size;
  const float* q_src  = (const float*)d_in[0];
  const float* kv_src = (const float*)d_in[1];
  const float* Wq = (const float*)d_in[2];  const float* bq = (const float*)d_in[3];
  const float* Wk = (const float*)d_in[4];  const float* bk = (const float*)d_in[5];
  const float* Wv = (const float*)d_in[6];  const float* bv = (const float*)d_in[7];
  const float* Wm = (const float*)d_in[8];
  const float* g1 = (const float*)d_in[9];  const float* b1 = (const float*)d_in[10];
  const float* W1 = (const float*)d_in[11]; const float* W2 = (const float*)d_in[12];
  const float* g2 = (const float*)d_in[13]; const float* b2 = (const float*)d_in[14];

  const size_t NLC  = (size_t)NB * LSEQ * DMODEL;   // 16,777,216
  const size_t NStC = (size_t)NB * STT * DMODEL;    // 33,554,432

  uint8_t* ws = (uint8_t*)d_ws;
  size_t off = 0;
  auto carve = [&](size_t bytes) -> void* {
    void* p = ws + off;
    off = (off + bytes + 255) & ~(size_t)255;
    return p;
  };
  bf16_t* kvbf = (bf16_t*)carve(NStC * 2);
  bf16_t* Kf   = (bf16_t*)carve(NStC * 2);       // directly after kvbf
  bf16_t* Vs   = (bf16_t*)carve(NStC * 2);
  bf16_t* qbf  = (bf16_t*)carve(NLC * 2);
  bf16_t* Qf   = (bf16_t*)carve(NLC * 2);
  bf16_t* Wqb  = (bf16_t*)carve((size_t)DMODEL * DMODEL * 2);
  bf16_t* Wkb  = (bf16_t*)carve((size_t)DMODEL * DMODEL * 2);
  bf16_t* Wvb  = (bf16_t*)carve((size_t)DMODEL * DMODEL * 2);
  bf16_t* Wmb  = (bf16_t*)carve((size_t)DMODEL * DMODEL * 2);
  bf16_t* W1b  = (bf16_t*)carve((size_t)DMODEL * DFF * 2);
  bf16_t* W2b  = (bf16_t*)carve((size_t)DFF * DMODEL * 2);
  float*  Ksum = (float*)carve((size_t)NB * NHEADS * HDIM * 4);
  bf16_t* KVb  = (bf16_t*)carve((size_t)NB * NHEADS * HDIM * HDIM * 2);
  float*  Zbuf = (float*)carve((size_t)NB * LSEQ * NHEADS * 4);
  float*  tmp  = (float*)carve(NLC * 4);
  float*  att1 = (float*)carve(NLC * 4);
  // lifetime-safe aliases
  bf16_t* hid  = kvbf;            // N*L*DFF bf16 == kvbf+Kf span, both dead
  bf16_t* atth = Vs;              // attention output reuses Vs
  bf16_t* a1bf = Qf;              // LN1 bf16 output reuses Qf
  float*  ff   = tmp;             // FFN2 output reuses merge tmp

  const dim3 blk(256);

  // 1) Precision conversions
  f2bf_kernel<<<(DMODEL * DMODEL + 255) / 256, blk, 0, stream>>>(Wq, Wqb, (size_t)DMODEL * DMODEL);
  f2bf_kernel<<<(DMODEL * DMODEL + 255) / 256, blk, 0, stream>>>(Wk, Wkb, (size_t)DMODEL * DMODEL);
  f2bf_kernel<<<(DMODEL * DMODEL + 255) / 256, blk, 0, stream>>>(Wv, Wvb, (size_t)DMODEL * DMODEL);
  f2bf_kernel<<<(DMODEL * DMODEL + 255) / 256, blk, 0, stream>>>(Wm, Wmb, (size_t)DMODEL * DMODEL);
  f2bf_kernel<<<(DMODEL * DFF + 255) / 256, blk, 0, stream>>>(W1, W1b, (size_t)DMODEL * DFF);
  f2bf_kernel<<<(DFF * DMODEL + 255) / 256, blk, 0, stream>>>(W2, W2b, (size_t)DFF * DMODEL);
  f2bf_kernel<<<(unsigned)(NLC / 256), blk, 0, stream>>>(q_src, qbf, NLC);
  kvcat_kernel<<<(unsigned)(NStC / 256), blk, 0, stream>>>(kv_src, q_src, kvbf);

  // 2) Projections (fused bias + elu+1 / scale), outputs bf16
  gemm_bf16_kernel<<<dim3(DMODEL / 128, (NB * LSEQ) / 128), blk, 0, stream>>>(
      qbf, Wqb, bq, Qf, NB * LSEQ, DMODEL, DMODEL, /*epi=*/1, 0.0f, /*bf16*/1);
  gemm_bf16_kernel<<<dim3(DMODEL / 128, (NB * STT) / 128), blk, 0, stream>>>(
      kvbf, Wkb, bk, Kf, NB * STT, DMODEL, DMODEL, /*epi=*/1, 0.0f, 1);
  gemm_bf16_kernel<<<dim3(DMODEL / 128, (NB * STT) / 128), blk, 0, stream>>>(
      kvbf, Wvb, bv, Vs, NB * STT, DMODEL, DMODEL, /*epi=*/2, 1.0f / (float)STT, 1);

  // 3) Linear attention core
  ksum_kernel<<<NB * NHEADS, blk, 0, stream>>>(Kf, Ksum);
  kv_accum_kernel<<<NB * NHEADS, blk, 0, stream>>>(Kf, Vs, KVb);
  z_kernel<<<(unsigned)((size_t)NB * LSEQ * NHEADS / 256), blk, 0, stream>>>(Qf, Ksum, Zbuf);
  attn_out_kernel<<<dim3(LSEQ / 128, NB * NHEADS), blk, 0, stream>>>(Qf, KVb, Zbuf, atth);

  // 4) Merge + LN1
  gemm_bf16_kernel<<<dim3(DMODEL / 128, (NB * LSEQ) / 128), blk, 0, stream>>>(
      atth, Wmb, nullptr, tmp, NB * LSEQ, DMODEL, DMODEL, /*epi=*/0, 0.0f, /*f32*/0);
  ln_kernel<<<NB * LSEQ, blk, 0, stream>>>(tmp, q_src, g1, b1, att1, a1bf);

  // 5) FFN + LN2 (final output into d_out)
  gemm_bf16_kernel<<<dim3(DFF / 128, (NB * LSEQ) / 128), blk, 0, stream>>>(
      a1bf, W1b, nullptr, hid, NB * LSEQ, DFF, DMODEL, /*epi=relu*/3, 0.0f, 1);
  gemm_bf16_kernel<<<dim3(DMODEL / 128, (NB * LSEQ) / 128), blk, 0, stream>>>(
      hid, W2b, nullptr, ff, NB * LSEQ, DMODEL, DFF, /*epi=*/0, 0.0f, 0);
  ln_kernel<<<NB * LSEQ, blk, 0, stream>>>(ff, att1, g2, b2, (float*)d_out, nullptr);

  // 6) Second tuple output: kv_src passthrough
  hipMemcpyAsync((float*)d_out + NLC, kv_src, NLC * sizeof(float),
                 hipMemcpyDeviceToDevice, stream);
}
